// Transformer_86131274154453
// MI455X (gfx1250) — matrix operand order repaired and checked by reading, not compiled, not run on hardware
//
#include <hip/hip_runtime.h>
#include <hip/hip_bf16.h>

// MI455X / gfx1250, wave32. All matmuls via V_WMMA_F32_16X16X32_F16.
//
// Shapes: B=4, N=1024, D=768, H=12, DH=64, F=3072, L=6.  M = B*N = 4096 rows.
// ~425 GFLOP total, weights ~170MB fp32 (~85MB as f16) -> compute bound:
// everything dense goes through WMMA with f16 operands / f32 accumulation.
// GEMM + attention inner loops are software-pipelined (register double
// buffering) so global_load_b128 latency overlaps the WMMAs instead of
// draining loadcnt before every matrix op.

#define BB   4
#define NSEQ 1024
#define DM   768
#define HH   12
#define DHD  64
#define FF   3072
#define LL   6
#define MROW (BB * NSEQ)   // 4096

typedef __attribute__((ext_vector_type(16))) _Float16 v16h;
typedef __attribute__((ext_vector_type(8)))  _Float16 h8;
typedef __attribute__((ext_vector_type(8)))  float    v8f;

union AU { v16h v; h8 h[2]; };

static __device__ inline v8f vzero8() {
  v8f z = {0.f, 0.f, 0.f, 0.f, 0.f, 0.f, 0.f, 0.f};
  return z;
}

static __device__ inline v8f wmma16(v16h a, v16h b, v8f c) {
  // D = A(16x32 f16) * B(32x16 f16) + C(16x16 f32)
  return __builtin_amdgcn_wmma_f32_16x16x32_f16(false, a, false, b, (short)0, c,
                                                false, false);
}

// ---------------------------------------------------------------------------
// Weight pack: fp32 row-major [K x Nc] -> f16 B-fragment tiles.
// Tile = 32(K) x 16(N). Linear tile id = ntile*KT + ktile. Within a tile,
// 32 lanes x 16 f16 each (32B, contiguous per lane):
//   lane<16 : K = ktile*32 + e      (e=0..15), N = ntile*16 + (lane&15)
//   lane>=16: K = ktile*32 + 16 + e,           N = ntile*16 + (lane&15)
// Contiguous k order (not AMD's table order); gemm16's A-frag loads use the same.
// ---------------------------------------------------------------------------
__global__ __launch_bounds__(256) void pack_w(const float* __restrict__ W,
                                              _Float16* __restrict__ Wp,
                                              int K, int Nc) {
  int idx  = blockIdx.x * 256 + threadIdx.x;     // one output f16 per thread
  int tile = idx >> 9;
  int rem  = idx & 511;
  int lane = rem >> 4;
  int e    = rem & 15;
  int KT   = K >> 5;
  int nt   = tile / KT;
  int kt   = tile - nt * KT;
  int kk   = kt * 32 + ((lane >= 16) ? 16 : 0) + e;
  int n    = nt * 16 + (lane & 15);
  Wp[idx]  = (_Float16)W[(size_t)kk * Nc + n];
}

// ---------------------------------------------------------------------------
// LayerNorm (eps = 1e-3, keras default) -> f16 row-major A matrix.
// One block per row of 768; 3 elements per thread.
// ---------------------------------------------------------------------------
__global__ __launch_bounds__(256) void ln_kernel(const float* __restrict__ X,
                                                 const float* __restrict__ g,
                                                 const float* __restrict__ bta,
                                                 _Float16* __restrict__ out) {
  __shared__ float sh[256];
  const int row = blockIdx.x;
  const int t   = threadIdx.x;
  const float* xr = X + (size_t)row * DM;
  float v0 = xr[t], v1 = xr[t + 256], v2 = xr[t + 512];
  sh[t] = v0 + v1 + v2;
  __syncthreads();
  for (int k = 128; k > 0; k >>= 1) {
    if (t < k) sh[t] += sh[t + k];
    __syncthreads();
  }
  float mean = sh[0] * (1.0f / DM);
  __syncthreads();
  float d0 = v0 - mean, d1 = v1 - mean, d2 = v2 - mean;
  sh[t] = d0 * d0 + d1 * d1 + d2 * d2;
  __syncthreads();
  for (int k = 128; k > 0; k >>= 1) {
    if (t < k) sh[t] += sh[t + k];
    __syncthreads();
  }
  float rs = rsqrtf(sh[0] * (1.0f / DM) + 1e-3f);
  _Float16* orow = out + (size_t)row * DM;
  orow[t]       = (_Float16)(d0 * rs * g[t]       + bta[t]);
  orow[t + 256] = (_Float16)(d1 * rs * g[t + 256] + bta[t + 256]);
  orow[t + 512] = (_Float16)(d2 * rs * g[t + 512] + bta[t + 512]);
}

// ---------------------------------------------------------------------------
// WMMA GEMM: C[4096 x Nc] = A[4096 x K](f16 row-major) * Wp(packed f16 tiles)
// 1 wave computes a 16 x 64 output block (4 accum tiles, A-frag reused 4x).
// Software-pipelined: stage kt+1's 10 b128 loads issue before stage kt's
// 4 WMMAs execute. blockDim=256 (8 waves). Epilogue modes:
//   0: QKV split  -> q16 (scaled by exp(temp)), k16 (row-major per head),
//                    v16t (transposed per head [DH x N])
//   1: X[row,col] += acc + bias[col]   (fp32 residual, Nc == 768)
//   2: outH = (f16) gelu_exact(acc + bias[col])
// ---------------------------------------------------------------------------
__global__ __launch_bounds__(256) void gemm16(
    const _Float16* __restrict__ A, const _Float16* __restrict__ Wp,
    int K, int Nc, int mode, const float* __restrict__ bias,
    float* __restrict__ X, _Float16* __restrict__ outH,
    const float* __restrict__ tempP, _Float16* __restrict__ q16,
    _Float16* __restrict__ k16, _Float16* __restrict__ v16t) {
  const int lane = threadIdx.x & 31;
  const int hf   = lane >> 4;       // half-wave select
  const int lo   = lane & 15;
  const int wid  = blockIdx.x * 8 + (threadIdx.x >> 5);
  const int nw   = Nc >> 6;         // N-groups of 64
  const int mt   = wid / nw;
  const int ntb  = (wid - mt * nw) << 2;
  const int KT   = K >> 5;          // 24 or 96 (always even)

  v8f acc[4];
#pragma unroll
  for (int j = 0; j < 4; ++j) acc[j] = vzero8();

  const _Float16* arow  = A + (size_t)(mt * 16 + lo) * K;
  const _Float16* wlane = Wp + (size_t)lane * 16;

  AU   a0, a1;
  v16h b0[4], b1[4];

#define LOAD_A(kt_, dst_)                                         \
  do {                                                            \
    const int k0_ = (kt_) << 5;                                   \
    (dst_).v = *(const v16h*)(arow + k0_ + hf * 16);              \
  } while (0)
#define LOAD_B(kt_, dst_)                                                     \
  do {                                                                        \
    _Pragma("unroll") for (int j_ = 0; j_ < 4; ++j_) {                        \
      (dst_)[j_] =                                                            \
          *(const v16h*)(wlane + ((size_t)(ntb + j_) * KT + (kt_)) * 512);    \
    }                                                                         \
  } while (0)

  LOAD_A(0, a0);
  LOAD_B(0, b0);
  for (int kt = 0; kt < KT; kt += 2) {
    // stage kt+1 loads issue before stage kt WMMAs
    LOAD_A(kt + 1, a1);
    LOAD_B(kt + 1, b1);
#pragma unroll
    for (int j = 0; j < 4; ++j) acc[j] = wmma16(a0.v, b0[j], acc[j]);
    const int kt2 = (kt + 2 < KT) ? kt + 2 : KT - 1;  // clamped dummy at end
    LOAD_A(kt2, a0);
    LOAD_B(kt2, b0);
#pragma unroll
    for (int j = 0; j < 4; ++j) acc[j] = wmma16(a1.v, b1[j], acc[j]);
  }
#undef LOAD_A
#undef LOAD_B

  const float scale = __expf(tempP[0]);  // only meaningful for mode 0
#pragma unroll
  for (int j = 0; j < 4; ++j) {
    const int col = (ntb + j) * 16 + lo;
#pragma unroll
    for (int r = 0; r < 8; ++r) {
      const int row = mt * 16 + r + hf * 8;  // C layout: M = r + 8*(lane/16)
      const float v = acc[j][r];
      if (mode == 0) {
        const int bi = row >> 10, n = row & (NSEQ - 1);
        if (col < DM) {
          const int h = col >> 6, dh = col & 63;
          q16[((size_t)(bi * HH + h) * NSEQ + n) * DHD + dh] =
              (_Float16)(v * scale);
        } else if (col < 2 * DM) {
          const int c = col - DM, h = c >> 6, dh = c & 63;
          k16[((size_t)(bi * HH + h) * NSEQ + n) * DHD + dh] = (_Float16)v;
        } else {
          const int c = col - 2 * DM, h = c >> 6, dh = c & 63;
          v16t[((size_t)(bi * HH + h) * DHD + dh) * NSEQ + n] = (_Float16)v;
        }
      } else if (mode == 1) {
        const size_t idx = (size_t)row * Nc + col;
        X[idx] = X[idx] + v + bias[col];
      } else {
        const float tv = v + bias[col];
        const float gl = 0.5f * tv * (1.0f + erff(tv * 0.70710678118654752f));
        outH[(size_t)row * Nc + col] = (_Float16)gl;
      }
    }
  }
}

// ---------------------------------------------------------------------------
// Flash attention with WMMA. Grid = B*H*(N/64) blocks, 128 threads (4 waves),
// each wave owns 16 query rows. Per 32-key block: 4 WMMAs for Q*K^T, online
// softmax (diag masked), P transposed via per-wave LDS, 4 WMMAs for P*V.
// Pipelining: next block's K fragments + this block's V fragments are issued
// before the softmax VALU work, which hides their latency.
// Output written as f16 row-major [4096 x 768] (A matrix of out-proj GEMM).
// ---------------------------------------------------------------------------
__global__ __launch_bounds__(128) void attn_kernel(
    const _Float16* __restrict__ q16, const _Float16* __restrict__ k16,
    const _Float16* __restrict__ v16t, _Float16* __restrict__ outA) {
  __shared__ __align__(64) _Float16 Pbuf[4 * 16 * 32];
  const int lane = threadIdx.x & 31;
  const int hf   = lane >> 4;
  const int lo   = lane & 15;
  const int w    = threadIdx.x >> 5;
  const int bh   = blockIdx.x >> 4;   // 0..47
  const int qblk = blockIdx.x & 15;
  const int q0   = qblk * 64 + w * 16;
  const int bi   = bh / HH, h = bh - bi * HH;

  const _Float16* qb = q16 + (size_t)bh * NSEQ * DHD;
  const _Float16* kb = k16 + (size_t)bh * NSEQ * DHD;
  const _Float16* vt = v16t + (size_t)bh * DHD * NSEQ;
  _Float16* Pw = Pbuf + w * 16 * 32;

  AU aq[2];  // Q A-fragments for K-chunks {0..31},{32..63}, reused all blocks
  {
    const _Float16* ar = qb + (size_t)(q0 + lo) * DHD;
#pragma unroll
    for (int kc = 0; kc < 2; ++kc) {
      aq[kc].v = *(const v16h*)(ar + kc * 32 + hf * 16);
    }
  }

  v8f o[4];
  float mr[8], lr[8];
#pragma unroll
  for (int t = 0; t < 4; ++t) o[t] = vzero8();
#pragma unroll
  for (int r = 0; r < 8; ++r) { mr[r] = -3.0e38f; lr[r] = 0.0f; }

  // K B-fragment double buffer: [tile t][k-chunk kc]
  v16h bk0[2][2], bk1[2][2];
#define LOAD_K(jb_, dst_)                                                \
  do {                                                                   \
    _Pragma("unroll") for (int t_ = 0; t_ < 2; ++t_) {                   \
      const _Float16* kr_ = kb + (size_t)((jb_) + t_ * 16 + lo) * DHD;   \
      _Pragma("unroll") for (int kc_ = 0; kc_ < 2; ++kc_) {              \
        (dst_)[t_][kc_] = *(const v16h*)(kr_ + kc_ * 32 + hf * 16);      \
      }                                                                  \
    }                                                                    \
  } while (0)

  LOAD_K(0, bk0);
  for (int jb = 0; jb < NSEQ; jb += 32) {
    // QK^T with current K fragments (already resident)
    v8f s[2];
    s[0] = vzero8();
    s[1] = vzero8();
#pragma unroll
    for (int t = 0; t < 2; ++t)
#pragma unroll
      for (int kc = 0; kc < 2; ++kc) s[t] = wmma16(aq[kc].v, bk0[t][kc], s[t]);

    // issue V fragments for this block + K fragments for the next block now;
    // the softmax VALU/trans work below hides their latency
    v16h bv[4];
#pragma unroll
    for (int t = 0; t < 4; ++t)
      bv[t] = *(const v16h*)(vt + (size_t)(t * 16 + lo) * NSEQ + jb + hf * 16);
    const int jbn = (jb + 32 < NSEQ) ? jb + 32 : jb;  // clamped dummy at end
    LOAD_K(jbn, bk1);

    // diagonal mask + online softmax (rows live across 16-lane halves)
#pragma unroll
    for (int r = 0; r < 8; ++r) {
      const int qg = q0 + r + hf * 8;
      if (qg == jb + lo)      s[0][r] = -3.0e38f;
      if (qg == jb + 16 + lo) s[1][r] = -3.0e38f;
      float rm = fmaxf(s[0][r], s[1][r]);
#pragma unroll
      for (int m = 1; m < 16; m <<= 1) rm = fmaxf(rm, __shfl_xor(rm, m, 32));
      const float mn = fmaxf(mr[r], rm);
      const float al = __expf(mr[r] - mn);
      mr[r] = mn;
      const float p0 = __expf(s[0][r] - mn);
      const float p1 = __expf(s[1][r] - mn);
      s[0][r] = p0;
      s[1][r] = p1;
      float rs = p0 + p1;
#pragma unroll
      for (int m = 1; m < 16; m <<= 1) rs += __shfl_xor(rs, m, 32);
      lr[r] = lr[r] * al + rs;
#pragma unroll
      for (int t = 0; t < 4; ++t) o[t][r] *= al;
    }

    // C-layout P -> LDS (row-major 16x32) -> reload as A-fragment
#pragma unroll
    for (int r = 0; r < 8; ++r) {
      Pw[(r + hf * 8) * 32 + lo]      = (_Float16)s[0][r];
      Pw[(r + hf * 8) * 32 + 16 + lo] = (_Float16)s[1][r];
    }
    AU ap;
    const _Float16* pr = Pw + lo * 32;
    ap.v = *(const v16h*)(pr + hf * 16);
#pragma unroll
    for (int t = 0; t < 4; ++t) o[t] = wmma16(ap.v, bv[t], o[t]);

    // rotate K double buffer
#pragma unroll
    for (int t = 0; t < 2; ++t)
#pragma unroll
      for (int kc = 0; kc < 2; ++kc) bk0[t][kc] = bk1[t][kc];
  }
#undef LOAD_K

#pragma unroll
  for (int r = 0; r < 8; ++r) {
    const float inv = 1.0f / lr[r];
    const int row   = bi * NSEQ + q0 + r + hf * 8;
#pragma unroll
    for (int t = 0; t < 4; ++t) {
      outA[(size_t)row * DM + h * 64 + t * 16 + lo] =
          (_Float16)(o[t][r] * inv);
    }
  }
}

// ---------------------------------------------------------------------------
extern "C" void kernel_launch(void* const* d_in, const int* in_sizes, int n_in,
                              void* d_out, int out_size, void* d_ws,
                              size_t ws_size, hipStream_t stream) {
  const float* x     = (const float*)d_in[0];
  const float* ln1_g = (const float*)d_in[1];
  const float* ln1_b = (const float*)d_in[2];
  const float* w_qkv = (const float*)d_in[3];
  const float* temp  = (const float*)d_in[4];
  const float* w_out = (const float*)d_in[5];
  const float* b_out = (const float*)d_in[6];
  const float* ln2_g = (const float*)d_in[7];
  const float* ln2_b = (const float*)d_in[8];
  const float* w1    = (const float*)d_in[9];
  const float* b1    = (const float*)d_in[10];
  const float* w2    = (const float*)d_in[11];
  const float* b2    = (const float*)d_in[12];

  char* wsc  = (char*)d_ws;
  size_t off = 0;
  auto alloc = [&](size_t bytes) -> char* {
    char* p = wsc + off;
    off     = (off + bytes + 255) & ~(size_t)255;
    return p;
  };
  const size_t E_WQKV = (size_t)DM * 3 * DM;     // 768*2304
  const size_t E_WOUT = (size_t)DM * DM;
  const size_t E_W1   = (size_t)DM * FF;
  const size_t E_W2   = (size_t)FF * DM;
  _Float16* Wqkv16 = (_Float16*)alloc(LL * E_WQKV * 2);
  _Float16* Wout16 = (_Float16*)alloc(LL * E_WOUT * 2);
  _Float16* W116   = (_Float16*)alloc(LL * E_W1 * 2);
  _Float16* W216   = (_Float16*)alloc(LL * E_W2 * 2);
  _Float16* LN16   = (_Float16*)alloc((size_t)MROW * DM * 2);
  _Float16* H16    = (_Float16*)alloc((size_t)MROW * FF * 2);
  _Float16* Q16    = (_Float16*)alloc((size_t)BB * HH * NSEQ * DHD * 2);
  _Float16* K16    = (_Float16*)alloc((size_t)BB * HH * NSEQ * DHD * 2);
  _Float16* V16T   = (_Float16*)alloc((size_t)BB * HH * NSEQ * DHD * 2);
  _Float16* AT16   = (_Float16*)alloc((size_t)MROW * DM * 2);

  // Pack all weights to f16 B-tile layout (re-done every call; deterministic).
  for (int i = 0; i < LL; ++i) {
    pack_w<<<(int)(E_WQKV / 256), 256, 0, stream>>>(
        w_qkv + (size_t)i * E_WQKV, Wqkv16 + (size_t)i * E_WQKV, DM, 3 * DM);
    pack_w<<<(int)(E_WOUT / 256), 256, 0, stream>>>(
        w_out + (size_t)i * E_WOUT, Wout16 + (size_t)i * E_WOUT, DM, DM);
    pack_w<<<(int)(E_W1 / 256), 256, 0, stream>>>(
        w1 + (size_t)i * E_W1, W116 + (size_t)i * E_W1, DM, FF);
    pack_w<<<(int)(E_W2 / 256), 256, 0, stream>>>(
        w2 + (size_t)i * E_W2, W216 + (size_t)i * E_W2, FF, DM);
  }

  // d_out (fp32 [4096 x 768]) is the persistent residual-stream buffer.
  float* X = (float*)d_out;
  hipMemcpyAsync(X, x, (size_t)MROW * DM * sizeof(float),
                 hipMemcpyDeviceToDevice, stream);

  // GEMM grids: blocks = (M/16) * (Nc/64) / 8 waves-per-block
  const int gQKV = (MROW / 16) * ((3 * DM) / 64) / 8;  // 1152
  const int gD   = (MROW / 16) * (DM / 64) / 8;        // 384
  const int gF   = (MROW / 16) * (FF / 64) / 8;        // 1536

  for (int i = 0; i < LL; ++i) {
    ln_kernel<<<MROW, 256, 0, stream>>>(X, ln1_g + i * DM, ln1_b + i * DM,
                                        LN16);
    gemm16<<<gQKV, 256, 0, stream>>>(LN16, Wqkv16 + (size_t)i * E_WQKV, DM,
                                     3 * DM, 0, b_out, X, H16, temp + i, Q16,
                                     K16, V16T);
    attn_kernel<<<BB * HH * (NSEQ / 64), 128, 0, stream>>>(Q16, K16, V16T,
                                                           AT16);
    gemm16<<<gD, 256, 0, stream>>>(AT16, Wout16 + (size_t)i * E_WOUT, DM, DM,
                                   1, b_out + i * DM, X, H16, temp, Q16, K16,
                                   V16T);
    ln_kernel<<<MROW, 256, 0, stream>>>(X, ln2_g + i * DM, ln2_b + i * DM,
                                        LN16);
    gemm16<<<gF, 256, 0, stream>>>(LN16, W116 + (size_t)i * E_W1, DM, FF, 2,
                                   b1 + i * FF, X, H16, temp, Q16, K16, V16T);
    gemm16<<<gD, 256, 0, stream>>>(H16, W216 + (size_t)i * E_W2, FF, DM, 1,
                                   b2 + i * DM, X, H16, temp, Q16, K16, V16T);
  }
}
